// Contrast_69380901699721
// MI455X (gfx1250) — matrix operand order repaired and checked by reading, not compiled, not run on hardware
//
#include <hip/hip_runtime.h>
#include <hip/hip_bf16.h>

#if __has_include(<hip/amd_detail/amd_gfx1250_TDM.h>)
#define TDM_6ARG 1
#endif

typedef __attribute__((ext_vector_type(16))) _Float16 v16h;
typedef __attribute__((ext_vector_type(8)))  float    v8f;
typedef __attribute__((ext_vector_type(4)))  unsigned int u32x4;
typedef __attribute__((ext_vector_type(8)))  int      i32x8;
typedef __attribute__((ext_vector_type(4)))  int      i32x4;

#define NROWS 8192          // 2*B
#define BHALF 4096          // B
#define DIM   512
#define DPAD  520           // +8 halves (16B) per-row LDS padding (1040B stride)
#define BM    128           // rows per block
#define BN    64            // cols per LDS tile
#define COLS_PER_BLOCK 1024 // column span per block (BN * 16 iterations)
#define NIT   (COLS_PER_BLOCK / BN)
#define INV_T 2.0f          // 1 / TEMPERATURE (0.5)

#define ZH_BYTES ((size_t)NROWS * DIM * 2)

// ---------------------------------------------------------------------------
// TDM: issue one tensor_load_to_lds moving a (rows x 512) f16 tile, row-major,
// into LDS with +16B padding after each 1024B row (=> DPAD row stride).
// D# packing per CDNA5 ISA.
// ---------------------------------------------------------------------------
__device__ __forceinline__ uint32_t lds_off(const void* p) {
    // LDS aperture: generic addr[31:0] == offset within wave's LDS allocation
    return (uint32_t)(uintptr_t)p;
}

__device__ __forceinline__ void tdm_load_tile(const _Float16* gsrc,
                                              uint32_t ldsByteOff, int rows) {
    const uint64_t ga = (uint64_t)(uintptr_t)gsrc;

    u32x4 g0;
    g0.x = 0x1u;                                   // count=1, user mode, no gather
    g0.y = ldsByteOff;                             // lds_addr
    g0.z = (uint32_t)ga;                           // global_addr[31:0]
    g0.w = (uint32_t)((ga >> 32) & 0x01FFFFFFu)    // global_addr[56:32]
         | (2u << 30);                             // type = 2 ("image")

    i32x8 g1;
    g1[0] = (1 << 16)          // data_size = 2 bytes
          | (1 << 20)          // pad_enable
          | (7 << 22)          // pad_interval: 256 DWORDs = 1024B
          | (3 << 25);         // pad_amount:   4 DWORDs = 16B
    g1[1] = (int)(DIM << 16);                               // tensor_dim0[15:0] @ bits 63:48
    g1[2] = (int)((NROWS & 0xFFFF) << 16);                  // tdim0 hi=0 | tensor_dim1 lo
    g1[3] = (int)(DIM << 16);                               // tdim1 hi=0 | tile_dim0=512
    g1[4] = rows;                                           // tile_dim1 (tile_dim2=0)
    g1[5] = DIM;                                            // tensor_dim0_stride[31:0]
    g1[6] = 0;
    g1[7] = 0;

    const i32x4 gz = {0, 0, 0, 0};
#if defined(TDM_6ARG)
    const i32x8 gz8 = {0, 0, 0, 0, 0, 0, 0, 0};
    __builtin_amdgcn_tensor_load_to_lds(g0, g1, gz, gz, gz8, 0);
#else
    __builtin_amdgcn_tensor_load_to_lds(g0, g1, gz, gz, 0);
#endif
}

// ---------------------------------------------------------------------------
// Kernel 1: L2-normalize rows of concat(x_i, x_j) -> f16 z, zero accumulators
// ---------------------------------------------------------------------------
__global__ __launch_bounds__(256)
void nxent_normalize(const float* __restrict__ xi,
                     const float* __restrict__ xj,
                     _Float16* __restrict__ zh,
                     float* __restrict__ rowSum,
                     float* __restrict__ diagV,
                     float* __restrict__ posV,
                     float* __restrict__ out)
{
    const int gid = blockIdx.x * 256 + threadIdx.x;
    if (gid < NROWS) { rowSum[gid] = 0.f; diagV[gid] = 0.f; posV[gid] = 0.f; }
    if (gid == 0) out[0] = 0.f;

    const int wave = threadIdx.x >> 5;   // 8 waves per block, one row each
    const int lane = threadIdx.x & 31;
    const int row  = blockIdx.x * 8 + wave;
    if (row >= NROWS) return;

    const float* __restrict__ src =
        (row < BHALF) ? (xi + (size_t)row * DIM)
                      : (xj + (size_t)(row - BHALF) * DIM);

    float4 v[4];
    float ss = 0.f;
#pragma unroll
    for (int c = 0; c < 4; ++c) {
        v[c] = *(const float4*)(src + c * 128 + lane * 4);
        ss += v[c].x * v[c].x + v[c].y * v[c].y + v[c].z * v[c].z + v[c].w * v[c].w;
    }
#pragma unroll
    for (int off = 16; off >= 1; off >>= 1)
        ss += __shfl_xor(ss, off, 32);

    const float scale = 1.f / fmaxf(sqrtf(ss), 1e-12f);

    _Float16* __restrict__ dst = zh + (size_t)row * DIM;
#pragma unroll
    for (int c = 0; c < 4; ++c) {
        union { _Float16 h[4]; ushort4 u; } p;
        p.h[0] = (_Float16)(v[c].x * scale);
        p.h[1] = (_Float16)(v[c].y * scale);
        p.h[2] = (_Float16)(v[c].z * scale);
        p.h[3] = (_Float16)(v[c].w * scale);
        *(ushort4*)(dst + c * 128 + lane * 4) = p.u;
    }
}

// ---------------------------------------------------------------------------
// Kernel 2: fused z@z^T tile + exp row-sum + diag/pos capture (WMMA f16)
// TDM double-buffered B tiles; A tile TDM-loaded once.
// grid: (NROWS/BM, NROWS/COLS_PER_BLOCK), block: 256 (8 waves, 2x2 reg tiles)
// ---------------------------------------------------------------------------
__global__ __launch_bounds__(256)
void nxent_sim(const _Float16* __restrict__ zh,
               float* __restrict__ rowSum,
               float* __restrict__ diagV,
               float* __restrict__ posV)
{
    extern __shared__ _Float16 smem[];
    _Float16* As  = smem;                        // BM x DPAD
    _Float16* Bs0 = smem + BM * DPAD;            // BN x DPAD (double buffer 0)
    _Float16* Bs1 = Bs0 + BN * DPAD;             // BN x DPAD (double buffer 1)

    const int tid    = threadIdx.x;
    const int rBase  = blockIdx.x * BM;
    const int cBlock = blockIdx.y * COLS_PER_BLOCK;
    const bool isW0  = (tid < 32);               // TDM issued once, by wave 0

    if (isW0) {
        tdm_load_tile(zh + (size_t)rBase  * DIM, lds_off(As),  BM);
        tdm_load_tile(zh + (size_t)cBlock * DIM, lds_off(Bs0), BN);
    }

    const int wave   = tid >> 5;
    const int lane   = tid & 31;
    const int wm     = wave & 3;       // 4 waves along M
    const int wn     = wave >> 2;      // 2 waves along N
    const int rowSub = wm * 32;
    const int colSub = wn * 32;
    const int l16    = lane & 15;
    const int hi     = lane >> 4;

    float sumP[2][8], diagP[2][8], posP[2][8];
#pragma unroll
    for (int mi = 0; mi < 2; ++mi)
#pragma unroll
        for (int v = 0; v < 8; ++v) { sumP[mi][v] = 0.f; diagP[mi][v] = 0.f; posP[mi][v] = 0.f; }

    const v8f zero8 = {0.f, 0.f, 0.f, 0.f, 0.f, 0.f, 0.f, 0.f};

    for (int it = 0; it < NIT; ++it) {
        const int cTile = cBlock + it * BN;

        // prefetch next B tile into the other buffer while we compute this one
        if (isW0) {
            if (it + 1 < NIT) {
                _Float16* nxt = ((it + 1) & 1) ? Bs1 : Bs0;
                tdm_load_tile(zh + (size_t)(cTile + BN) * DIM, lds_off(nxt), BN);
                __builtin_amdgcn_s_wait_tensorcnt(1);  // A + B(it) resident
            } else {
                __builtin_amdgcn_s_wait_tensorcnt(0);  // everything resident
            }
        }
        __syncthreads();   // publish TDM-written LDS to all waves

        const _Float16* Bcur = (it & 1) ? Bs1 : Bs0;

        v8f acc[2][2];
        acc[0][0] = zero8; acc[0][1] = zero8; acc[1][0] = zero8; acc[1][1] = zero8;

#pragma unroll
        for (int ks = 0; ks < DIM / 32; ++ks) {
            const int k0 = ks * 32;

            // A fragments: ISA 16-bit 16x32 layout — lane m=l16, two 16B chunks
            v16h a[2];
#pragma unroll
            for (int mi = 0; mi < 2; ++mi) {
                const _Float16* p = As + (rowSub + mi * 16 + l16) * DPAD + k0 + hi * 8;
                union { v16h h; uint4 u[2]; } t;
                t.u[0] = *(const uint4*)p;
                t.u[1] = *(const uint4*)(p + 16);
                a[mi] = t.h;
            }
            // B fragments: column n=l16, two 16B chunks at k0 + hi*8 and k0 + 16 + hi*8
            v16h b[2];
#pragma unroll
            for (int ni = 0; ni < 2; ++ni) {
                const _Float16* p = Bcur + (colSub + ni * 16 + l16) * DPAD + k0 + hi * 8;
                union { v16h h; uint4 u[2]; } t;
                t.u[0] = *(const uint4*)p;
                t.u[1] = *(const uint4*)(p + 16);
                b[ni] = t.h;
            }
#pragma unroll
            for (int mi = 0; mi < 2; ++mi)
#pragma unroll
                for (int ni = 0; ni < 2; ++ni)
                    acc[mi][ni] = __builtin_amdgcn_wmma_f32_16x16x32_f16(
                        false, a[mi], false, b[ni], (short)0, acc[mi][ni], false, false);
        }

        // ---- fused epilogue: exp/rowsum + diag & positive-pair capture ----
#pragma unroll
        for (int mi = 0; mi < 2; ++mi) {
#pragma unroll
            for (int ni = 0; ni < 2; ++ni) {
#pragma unroll
                for (int v = 0; v < 8; ++v) {
                    const float s = acc[mi][ni][v];
                    const int r = rBase + rowSub + mi * 16 + v + hi * 8;
                    const int c = cTile + colSub + ni * 16 + l16;
                    sumP[mi][v] += __expf(s * INV_T);
                    if (c == r) diagP[mi][v] += s;
                    const int partner = (r < BHALF) ? (r + BHALF) : (r - BHALF);
                    if (c == partner) posP[mi][v] += s;
                }
            }
        }
        __syncthreads();   // compute(it) done before buffer it&1 is refilled
    }

    // ---- reduce across the 16 lanes sharing each output row ----
#pragma unroll
    for (int mi = 0; mi < 2; ++mi) {
#pragma unroll
        for (int v = 0; v < 8; ++v) {
            float s = sumP[mi][v], d = diagP[mi][v], p = posP[mi][v];
#pragma unroll
            for (int off = 8; off >= 1; off >>= 1) {
                s += __shfl_xor(s, off, 16);
                d += __shfl_xor(d, off, 16);
                p += __shfl_xor(p, off, 16);
            }
            if (l16 == 0) {
                const int r = rBase + rowSub + mi * 16 + v + hi * 8;
                atomicAdd(&rowSum[r], s);
                if (d != 0.f) atomicAdd(&diagV[r], d);
                if (p != 0.f) atomicAdd(&posV[r], p);
            }
        }
    }
}

// ---------------------------------------------------------------------------
// Kernel 3: per-row loss -> mean
// ---------------------------------------------------------------------------
__global__ __launch_bounds__(256)
void nxent_finalize(const float* __restrict__ rowSum,
                    const float* __restrict__ diagV,
                    const float* __restrict__ posV,
                    float* __restrict__ out)
{
    const int r = blockIdx.x * 256 + threadIdx.x;
    if (r >= NROWS) return;
    const float denom = rowSum[r] - __expf(diagV[r] * INV_T);
    const float loss  = __logf(denom) - posV[r] * INV_T;
    atomicAdd(out, loss * (1.0f / NROWS));
}

// ---------------------------------------------------------------------------
extern "C" void kernel_launch(void* const* d_in, const int* in_sizes, int n_in,
                              void* d_out, int out_size, void* d_ws, size_t ws_size,
                              hipStream_t stream) {
    (void)in_sizes; (void)n_in; (void)out_size; (void)ws_size;

    const float* xi = (const float*)d_in[0];
    const float* xj = (const float*)d_in[1];
    float* out = (float*)d_out;

    char* ws = (char*)d_ws;
    _Float16* zh   = (_Float16*)ws;
    float* rowSum  = (float*)(ws + ZH_BYTES);
    float* diagV   = rowSum + NROWS;
    float* posV    = diagV  + NROWS;

    const size_t smemBytes = (size_t)(BM + 2 * BN) * DPAD * sizeof(_Float16); // 266,240 B
    static bool attrSet = false;   // host-side config, deterministic & capture-safe
    if (!attrSet) {
        (void)hipFuncSetAttribute((const void*)nxent_sim,
                                  hipFuncAttributeMaxDynamicSharedMemorySize,
                                  (int)smemBytes);
        attrSet = true;
    }

    nxent_normalize<<<NROWS / 8, 256, 0, stream>>>(xi, xj, zh, rowSum, diagV, posV, out);

    dim3 grid(NROWS / BM, NROWS / COLS_PER_BLOCK);  // 64 x 8 = 512 blocks
    nxent_sim<<<grid, 256, smemBytes, stream>>>(zh, rowSum, diagV, posV);

    nxent_finalize<<<NROWS / 256, 256, 0, stream>>>(rowSum, diagV, posV, out);
}
